// SelfAttentionMulti_77541339562067
// MI455X (gfx1250) — compile-verified, matrix operand order checked by reading, not run on hardware
//
#include <hip/hip_runtime.h>

#define SEQL   2048
#define NBATCH 2
#define DIM    1024
#define NHEADS 16
#define DHEAD  64

typedef __attribute__((ext_vector_type(16))) _Float16 v16h;
typedef __attribute__((ext_vector_type(8)))  float    v8f;

union FragU { v16h v; uint4 u[2]; };

__device__ __forceinline__ v8f wmma_f16(v16h a, v16h b, v8f c) {
  // D = A(16x32) * B(32x16) + C, f32 accumulate
  return __builtin_amdgcn_wmma_f32_16x16x32_f16(false, a, false, b,
                                                (short)0, c, false, false);
}

// Load a 16x32 f16 fragment from a row-major matrix, rows at stride `ld`.
// Per CDNA5 ISA 16-bit A/B^T layout: lane r (0..15) -> row r;
// lanes 0..15 hold K 0..7 and 16..23, lanes 16..31 hold K 8..15 and 24..31.
// Both halves are contiguous 16-byte chunks -> two b128 loads per lane.
__device__ __forceinline__ v16h load_frag(const _Float16* p, int ld) {
  const int lane = threadIdx.x & 31;
  const int r  = lane & 15;
  const int hh = lane >> 4;
  const _Float16* q = p + (size_t)r * ld + hh * 8;
  FragU f;
  f.u[0] = *(const uint4*)(q);
  f.u[1] = *(const uint4*)(q + 16);
  return f.v;
}

// ---------------- fp32 -> f16 conversion ----------------
__global__ void cvt_f32_to_f16(const float* __restrict__ s,
                               _Float16* __restrict__ d, int n) {
  int i = (blockIdx.x * blockDim.x + threadIdx.x) * 4;
  if (i < n) {
    float4 x = *(const float4*)(s + i);
    d[i + 0] = (_Float16)x.x;
    d[i + 1] = (_Float16)x.y;
    d[i + 2] = (_Float16)x.z;
    d[i + 3] = (_Float16)x.w;
  }
}

// ================= GEMM: C = A(MxK) @ B(NxK)^T =================
// block = 256 threads = 8 waves (2 x 4); block tile 64 x 256.
// Each wave: 32 x 64 (2 A-frags x 4 B-frags = 8 WMMAs per 32-wide k-step),
// with software-pipelined fragment loads (next k-step issued before MMAs).

#define GEMM_BODY(EPILOGUE)                                                    \
  const int w    = threadIdx.x >> 5;                                           \
  const int lane = threadIdx.x & 31;                                           \
  const int row0 = blockIdx.y * 64 + (w >> 2) * 32;                            \
  const int col0 = blockIdx.x * 256 + (w & 3) * 64;                            \
  const _Float16* Ap = A + (size_t)row0 * K;                                   \
  const _Float16* Bp = B + (size_t)col0 * K;                                   \
  v8f acc[2][4] = {};                                                          \
  v16h a0 = load_frag(Ap, K);                                                  \
  v16h a1 = load_frag(Ap + (size_t)16 * K, K);                                 \
  v16h b0 = load_frag(Bp, K);                                                  \
  v16h b1 = load_frag(Bp + (size_t)16 * K, K);                                 \
  v16h b2 = load_frag(Bp + (size_t)32 * K, K);                                 \
  v16h b3 = load_frag(Bp + (size_t)48 * K, K);                                 \
  for (int k0 = 32; k0 < K; k0 += 32) {                                        \
    __builtin_prefetch(Ap + k0 + 32, 0, 3);  /* speculative, safe past end */  \
    __builtin_prefetch(Bp + k0 + 32, 0, 3);                                    \
    v16h na0 = load_frag(Ap + k0, K);                                          \
    v16h na1 = load_frag(Ap + (size_t)16 * K + k0, K);                         \
    v16h nb0 = load_frag(Bp + k0, K);                                          \
    v16h nb1 = load_frag(Bp + (size_t)16 * K + k0, K);                         \
    v16h nb2 = load_frag(Bp + (size_t)32 * K + k0, K);                         \
    v16h nb3 = load_frag(Bp + (size_t)48 * K + k0, K);                         \
    acc[0][0] = wmma_f16(a0, b0, acc[0][0]);                                   \
    acc[0][1] = wmma_f16(a0, b1, acc[0][1]);                                   \
    acc[0][2] = wmma_f16(a0, b2, acc[0][2]);                                   \
    acc[0][3] = wmma_f16(a0, b3, acc[0][3]);                                   \
    acc[1][0] = wmma_f16(a1, b0, acc[1][0]);                                   \
    acc[1][1] = wmma_f16(a1, b1, acc[1][1]);                                   \
    acc[1][2] = wmma_f16(a1, b2, acc[1][2]);                                   \
    acc[1][3] = wmma_f16(a1, b3, acc[1][3]);                                   \
    a0 = na0; a1 = na1;                                                        \
    b0 = nb0; b1 = nb1; b2 = nb2; b3 = nb3;                                    \
  }                                                                            \
  acc[0][0] = wmma_f16(a0, b0, acc[0][0]);                                     \
  acc[0][1] = wmma_f16(a0, b1, acc[0][1]);                                     \
  acc[0][2] = wmma_f16(a0, b2, acc[0][2]);                                     \
  acc[0][3] = wmma_f16(a0, b3, acc[0][3]);                                     \
  acc[1][0] = wmma_f16(a1, b0, acc[1][0]);                                     \
  acc[1][1] = wmma_f16(a1, b1, acc[1][1]);                                     \
  acc[1][2] = wmma_f16(a1, b2, acc[1][2]);                                     \
  acc[1][3] = wmma_f16(a1, b3, acc[1][3]);                                     \
  const int r = lane & 15, hh = lane >> 4;                                     \
  for (int i2 = 0; i2 < 2; ++i2)                                               \
    for (int j = 0; j < 4; ++j)                                                \
      for (int i = 0; i < 8; ++i) {                                            \
        int row = row0 + 16 * i2 + i + 8 * hh;                                 \
        int col = col0 + 16 * j + r;                                           \
        EPILOGUE;                                                              \
      }

__global__ __launch_bounds__(256) void gemm_abt_f16(
    const _Float16* __restrict__ A, const _Float16* __restrict__ B,
    _Float16* __restrict__ C, int M, int N, int K) {
  GEMM_BODY(C[(size_t)row * N + col] = (_Float16)acc[i2][j][i])
}

__global__ __launch_bounds__(256) void gemm_abt_f32_bias(
    const _Float16* __restrict__ A, const _Float16* __restrict__ B,
    float* __restrict__ C, const float* __restrict__ bias,
    int M, int N, int K) {
  GEMM_BODY(C[(size_t)row * N + col] = acc[i2][j][i] + bias[col])
}

// ---------------- causal flash attention ----------------
// grid: (SEQL/128, NHEADS, NBATCH); block 256 = 8 waves.
// Each block: 128 query rows; wave w owns 16 rows. Key blocks of 64.
__global__ __launch_bounds__(256) void attn_kernel(
    const _Float16* __restrict__ Q, const _Float16* __restrict__ Kh,
    const _Float16* __restrict__ Vh, _Float16* __restrict__ O) {
  __shared__ alignas(16) _Float16 Kt[64][72];      // [key][d]  (padded)
  __shared__ alignas(16) _Float16 Vt[64][72];      // [d][key]  (transposed)
  __shared__ alignas(16) _Float16 Pw[8][16][72];   // per-wave P tiles

  const int w    = threadIdx.x >> 5;
  const int lane = threadIdx.x & 31;
  const int r = lane & 15, hh = lane >> 4;
  const int qbase = blockIdx.x * 128;
  const int h = blockIdx.y;
  const int b = blockIdx.z;
  const size_t rowbase = (size_t)b * SEQL;
  const int qr0 = qbase + 16 * w;

  v8f acc[4] = {};
  float m_i[8], l_i[8];
  for (int i = 0; i < 8; ++i) { m_i[i] = -3.0e38f; l_i[i] = 0.0f; }

  const int nkb = qbase / 64 + 2;   // causal: keys up to qbase+127
  for (int kb = 0; kb < nkb; ++kb) {
    const int key0 = kb * 64;
    __syncthreads();
    // stage K tile (64 keys x 64 d) via async global->LDS DMA (ASYNCcnt path)
    for (int it = threadIdx.x; it < 512; it += 256) {
      int kr = it >> 3, c8 = (it & 7) * 8;
      unsigned ldsoff = (unsigned)(size_t)(&Kt[kr][c8]);   // LDS byte offset
      const _Float16* gp =
          Kh + (rowbase + key0 + kr) * DIM + h * DHEAD + c8;
      asm volatile("global_load_async_to_lds_b128 %0, %1, off"
                   :: "v"(ldsoff), "v"(gp)
                   : "memory");
    }
    // stage V transposed: Vt[d][key] (DMA cannot transpose -> ds stores)
    for (int it = threadIdx.x; it < 4096; it += 256) {
      int kr = it >> 6, c = it & 63;
      Vt[c][kr] = Vh[(rowbase + key0 + kr) * DIM + h * DHEAD + c];
    }
    asm volatile("s_wait_asynccnt 0" ::: "memory");
    __syncthreads();

    if (key0 <= qr0 + 15) {   // wave-uniform: skip fully-masked key blocks
      // ---- S = Q @ K^T  (16 x 64 scores per wave) ----
      v16h qf0 = load_frag(Q + (rowbase + qr0) * DIM + h * DHEAD, DIM);
      v16h qf1 = load_frag(Q + (rowbase + qr0) * DIM + h * DHEAD + 32, DIM);
      v8f s[4] = {};
      for (int t = 0; t < 4; ++t) {
        v16h k0f = load_frag(&Kt[16 * t][0], 72);
        v16h k1f = load_frag(&Kt[16 * t][32], 72);
        s[t] = wmma_f16(qf0, k0f, s[t]);
        s[t] = wmma_f16(qf1, k1f, s[t]);
      }
      // ---- scale + causal mask ----
      for (int t = 0; t < 4; ++t)
        for (int i = 0; i < 8; ++i) {
          float v = s[t][i] * 0.125f;              // 1/sqrt(64)
          int kidx = key0 + 16 * t + r;
          int qidx = qr0 + i + 8 * hh;
          s[t][i] = (kidx > qidx) ? -3.0e38f : v;
        }
      // ---- online softmax update ----
      float p[4][8];
      for (int i = 0; i < 8; ++i) {
        float mx = s[0][i];
        for (int t = 1; t < 4; ++t) mx = fmaxf(mx, s[t][i]);
        for (int off = 1; off < 16; off <<= 1)
          mx = fmaxf(mx, __shfl_xor(mx, off, 32));
        float mnew  = fmaxf(m_i[i], mx);
        float alpha = __expf(m_i[i] - mnew);
        float rs = 0.0f;
        for (int t = 0; t < 4; ++t) {
          p[t][i] = __expf(s[t][i] - mnew);
          rs += p[t][i];
        }
        for (int off = 1; off < 16; off <<= 1)
          rs += __shfl_xor(rs, off, 32);
        l_i[i] = l_i[i] * alpha + rs;
        m_i[i] = mnew;
        for (int t = 0; t < 4; ++t) acc[t][i] *= alpha;
      }
      // ---- P -> LDS (C-layout -> A-layout round trip) ----
      for (int t = 0; t < 4; ++t)
        for (int i = 0; i < 8; ++i)
          Pw[w][i + 8 * hh][16 * t + r] = (_Float16)p[t][i];
      // ---- acc += P @ V  (B = Vt, both row-major) ----
      v16h pf0 = load_frag(&Pw[w][0][0], 72);
      v16h pf1 = load_frag(&Pw[w][0][32], 72);
      for (int t = 0; t < 4; ++t) {
        v16h v0 = load_frag(&Vt[16 * t][0], 72);
        v16h v1 = load_frag(&Vt[16 * t][32], 72);
        acc[t] = wmma_f16(pf0, v0, acc[t]);
        acc[t] = wmma_f16(pf1, v1, acc[t]);
      }
    }
  }
  // ---- normalize and store O (f16) ----
  for (int t = 0; t < 4; ++t)
    for (int i = 0; i < 8; ++i) {
      int row = qr0 + i + 8 * hh;
      O[(rowbase + row) * DIM + h * DHEAD + 16 * t + r] =
          (_Float16)(acc[t][i] / l_i[i]);
    }
}

extern "C" void kernel_launch(void* const* d_in, const int* in_sizes, int n_in,
                              void* d_out, int out_size, void* d_ws, size_t ws_size,
                              hipStream_t stream) {
  (void)in_sizes; (void)n_in; (void)out_size; (void)ws_size;
  const float* X  = (const float*)d_in[0];
  const float* Wk = (const float*)d_in[1];
  const float* Wq = (const float*)d_in[2];
  const float* Wv = (const float*)d_in[3];
  const float* Wp = (const float*)d_in[4];
  const float* bp = (const float*)d_in[5];

  const size_t MTOK = (size_t)NBATCH * SEQL;   // 4096 token rows
  const size_t SZX  = MTOK * DIM;              // 4M elements
  const size_t SZW  = (size_t)DIM * DIM;       // 1M elements

  _Float16* ws  = (_Float16*)d_ws;
  _Float16* Xh  = ws;
  _Float16* Wqh = Xh  + SZX;
  _Float16* Wkh = Wqh + SZW;
  _Float16* Wvh = Wkh + SZW;
  _Float16* Wph = Wvh + SZW;
  _Float16* Qh  = Wph + SZW;
  _Float16* Kh  = Qh  + SZX;
  _Float16* Vh  = Kh  + SZX;
  _Float16* Oh  = Vh  + SZX;   // total ~48 MB of f16 workspace

  // 1) fp32 -> f16
  cvt_f32_to_f16<<<(int)(SZX / 1024), 256, 0, stream>>>(X,  Xh,  (int)SZX);
  cvt_f32_to_f16<<<(int)(SZW / 1024), 256, 0, stream>>>(Wq, Wqh, (int)SZW);
  cvt_f32_to_f16<<<(int)(SZW / 1024), 256, 0, stream>>>(Wk, Wkh, (int)SZW);
  cvt_f32_to_f16<<<(int)(SZW / 1024), 256, 0, stream>>>(Wv, Wvh, (int)SZW);
  cvt_f32_to_f16<<<(int)(SZW / 1024), 256, 0, stream>>>(Wp, Wph, (int)SZW);

  // 2) QKV projections: y = X @ W^T
  dim3 ggrid(DIM / 256, (unsigned)(MTOK / 64));
  gemm_abt_f16<<<ggrid, 256, 0, stream>>>(Xh, Wqh, Qh, (int)MTOK, DIM, DIM);
  gemm_abt_f16<<<ggrid, 256, 0, stream>>>(Xh, Wkh, Kh, (int)MTOK, DIM, DIM);
  gemm_abt_f16<<<ggrid, 256, 0, stream>>>(Xh, Wvh, Vh, (int)MTOK, DIM, DIM);

  // 3) causal flash attention
  dim3 agrid(SEQL / 128, NHEADS, NBATCH);
  attn_kernel<<<agrid, 256, 0, stream>>>(Qh, Kh, Vh, Oh);

  // 4) output projection (+bias), f32 out
  gemm_abt_f32_bias<<<ggrid, 256, 0, stream>>>(Oh, Wph, (float*)d_out, bp,
                                               (int)MTOK, DIM, DIM);
}
